// ResidualAttentionBlock_8495445311913
// MI455X (gfx1250) — compile-verified, matrix operand order checked by reading, not run on hardware
//
#include <hip/hip_runtime.h>
#include <hip/hip_bf16.h>
#include <math.h>

#define NB 8
#define LSEQ 512
#define DMODEL 768
#define NH 12
#define HDIM 64
#define MROWS (NB * LSEQ) /* 4096 */

typedef __attribute__((ext_vector_type(16))) _Float16 v16h;
typedef __attribute__((ext_vector_type(8)))  float    v8f;
typedef __attribute__((ext_vector_type(4)))  float    f32x4;

union FragU { v16h v; f32x4 q[2]; };

// Per-lane fragment load: 16 halfs = two 16B chunks at +0 and +16 halfs.
__device__ __forceinline__ v16h ld_frag(const _Float16* p) {
  FragU f;
  f.q[0] = *(const f32x4*)(p);
  f.q[1] = *(const f32x4*)(p + 16);
  return f.v;
}

__device__ __forceinline__ v8f wmma16(v16h a, v16h b, v8f c) {
  // (neg_a, A, neg_b, B, c_mod, C, reuse_a, reuse_b)
  return __builtin_amdgcn_wmma_f32_16x16x32_f16(false, a, false, b, (short)0, c,
                                                false, false);
}

// Hardware reciprocal: single v_rcp_f32 (no IEEE div_scale/div_fmas chain).
__device__ __forceinline__ float fast_rcp(float x) {
  return __builtin_amdgcn_rcpf(x);
}

// Branch-free tanh from the hardware exp/rcp units: t=exp(-2|x|), r=(1-t)/(1+t)
__device__ __forceinline__ float fast_tanh(float x) {
  float t = __expf(-2.0f * fabsf(x));
  float r = (1.0f - t) * fast_rcp(1.0f + t);
  return copysignf(r, x);
}

// ---------------------------------------------------------------------------
// Elementwise helpers
// ---------------------------------------------------------------------------
__global__ __launch_bounds__(256) void cvt_f16_kernel(const float* __restrict__ in,
                                                      _Float16* __restrict__ out,
                                                      int n) {
  int i = blockIdx.x * 256 + threadIdx.x;
  if (i < n) out[i] = (_Float16)in[i];
}

// k: (N,L,H,HD) fp32 -> kt: [n][h][l][hd] f16
__global__ __launch_bounds__(256) void permute_k_kernel(const float* __restrict__ in,
                                                        _Float16* __restrict__ out) {
  int i = blockIdx.x * 256 + threadIdx.x; // NB*LSEQ*NH*HDIM = 3145728
  int c = i & (HDIM - 1);
  int h = (i >> 6) % NH;
  int l = ((i >> 6) / NH) % LSEQ;
  int n = i / (HDIM * NH * LSEQ);
  out[(((size_t)(n * NH + h) * LSEQ) + l) * HDIM + c] = (_Float16)in[i];
}

// v: (N,L,H,HD) fp32 -> vt: [n][h][hd][l] f16  (transposed for PV B-frags)
__global__ __launch_bounds__(256) void permute_v_kernel(const float* __restrict__ in,
                                                        _Float16* __restrict__ out) {
  int i = blockIdx.x * 256 + threadIdx.x;
  int c = i & (HDIM - 1);
  int h = (i >> 6) % NH;
  int l = ((i >> 6) / NH) % LSEQ;
  int n = i / (HDIM * NH * LSEQ);
  out[((size_t)(n * NH + h) * HDIM + c) * LSEQ + l] = (_Float16)in[i];
}

// LayerNorm over D=768, one 256-thread block per row, f16 output.
__global__ __launch_bounds__(256) void ln_f16_kernel(const float* __restrict__ x,
                                                     const float* __restrict__ w,
                                                     const float* __restrict__ b,
                                                     _Float16* __restrict__ out) {
  int row = blockIdx.x;
  int tid = threadIdx.x;
  const float* xr = x + (size_t)row * DMODEL;
  float v0 = xr[tid], v1 = xr[tid + 256], v2 = xr[tid + 512];
  __shared__ float rs[256], rs2[256];
  rs[tid]  = v0 + v1 + v2;
  rs2[tid] = v0 * v0 + v1 * v1 + v2 * v2;
  __syncthreads();
  for (int off = 128; off > 0; off >>= 1) {
    if (tid < off) { rs[tid] += rs[tid + off]; rs2[tid] += rs2[tid + off]; }
    __syncthreads();
  }
  float mu  = rs[0] * (1.0f / DMODEL);
  float var = rs2[0] * (1.0f / DMODEL) - mu * mu;
  float inv = rsqrtf(var + 1e-5f);
  _Float16* o = out + (size_t)row * DMODEL;
  o[tid]       = (_Float16)((v0 - mu) * inv * w[tid]       + b[tid]);
  o[tid + 256] = (_Float16)((v1 - mu) * inv * w[tid + 256] + b[tid + 256]);
  o[tid + 512] = (_Float16)((v2 - mu) * inv * w[tid + 512] + b[tid + 512]);
}

// gate[n,h,l] = m[n,l] ? 2*sigmoid(-sum_c |q2-k|/8) : 0
__global__ __launch_bounds__(256) void gate_kernel(const _Float16* __restrict__ qpm,
                                                   const _Float16* __restrict__ kt,
                                                   const unsigned char* __restrict__ msk,
                                                   float* __restrict__ gate) {
  int i = blockIdx.x * 256 + threadIdx.x; // NB*NH*LSEQ = 49152
  if (i >= NB * NH * LSEQ) return;
  int l = i & (LSEQ - 1);
  int h = (i >> 9) % NH;
  int n = i / (LSEQ * NH);
  const _Float16* q2 = qpm + (size_t)(n * LSEQ + l) * (2 * DMODEL) + h * 128 + 64;
  const _Float16* kp = kt + ((size_t)(n * NH + h) * LSEQ + l) * HDIM;
  float s = 0.f;
  for (int c = 0; c < HDIM; ++c) s += fabsf((float)q2[c] - (float)kp[c]);
  float g = 2.0f * fast_rcp(1.0f + __expf(s * 0.125f));
  gate[i] = msk[n * LSEQ + l] ? g : 0.0f;
}

// ---------------------------------------------------------------------------
// WMMA GEMM: C(M x Nc) = A(M x K, f16) * B(Nc x K, f16)^T + bias, epilogues:
//   EP=0: f16 out (acc+bias)                       [qp]
//   EP=1: f32 out (acc+bias+resid)                 [out-proj, final proj]
//   EP=2: f16 out silu-ish: h*sigmoid(1.702h)      [fc]
// Block = 256 thr = 8 waves as 2(M) x 4(N); each wave does a 32x64 strip
// (2 A-frags + 4 B-frags -> 8 WMMA per 32-wide k-step).
// WG tile = 64(M) x 256(N). All problem dims divide exactly.
// ---------------------------------------------------------------------------
template <int EP>
__global__ __launch_bounds__(256) void gemm_f16w(const _Float16* __restrict__ A, int lda,
                                                 const _Float16* __restrict__ B, int ldb,
                                                 const float* __restrict__ bias,
                                                 const float* __restrict__ resid, int ldr,
                                                 void* __restrict__ Cout, int ldc, int K) {
  int tid  = threadIdx.x;
  int wave = tid >> 5;
  int lane = tid & 31;
  int wm = wave & 1, wn = wave >> 1;
  int m0 = blockIdx.y * 64 + wm * 32;
  int n0 = blockIdx.x * 256 + wn * 64;
  int lr    = lane & 15;
  int koff  = (lane < 16) ? 0 : 8;
  int rbase = (lane < 16) ? 0 : 8;

  const _Float16* Ap = A + (size_t)(m0 + lr) * lda + koff;
  const _Float16* Bp = B + (size_t)(n0 + lr) * ldb + koff;

  v8f acc0[4] = {v8f{}, v8f{}, v8f{}, v8f{}};
  v8f acc1[4] = {v8f{}, v8f{}, v8f{}, v8f{}};
  for (int k0 = 0; k0 < K; k0 += 32) {
    v16h a0 = ld_frag(Ap + k0);
    v16h a1 = ld_frag(Ap + (size_t)16 * lda + k0);
    __builtin_prefetch(Ap + k0 + 32, 0, 1);   // next k-step (global_prefetch_b8)
    __builtin_prefetch(Bp + k0 + 32, 0, 1);
#pragma unroll
    for (int nt = 0; nt < 4; ++nt) {
      v16h b = ld_frag(Bp + (size_t)nt * 16 * ldb + k0);
      acc0[nt] = wmma16(a0, b, acc0[nt]);
      acc1[nt] = wmma16(a1, b, acc1[nt]);
    }
  }

#pragma unroll
  for (int nt = 0; nt < 4; ++nt) {
    int col  = n0 + nt * 16 + lr;
    float bv = bias[col];
#pragma unroll
    for (int mt = 0; mt < 2; ++mt) {
#pragma unroll
      for (int r8 = 0; r8 < 8; ++r8) {
        int row = m0 + mt * 16 + rbase + r8;
        float v = (mt == 0 ? acc0[nt][r8] : acc1[nt][r8]) + bv;
        if (EP == 0) {
          ((_Float16*)Cout)[(size_t)row * ldc + col] = (_Float16)v;
        } else if (EP == 1) {
          ((float*)Cout)[(size_t)row * ldc + col] = v + resid[(size_t)row * ldr + col];
        } else {
          float sv = v * fast_rcp(1.0f + __expf(-1.702f * v));
          ((_Float16*)Cout)[(size_t)row * ldc + col] = (_Float16)sv;
        }
      }
    }
  }
}

// ---------------------------------------------------------------------------
// Attention: one block per (n, h, 32-query slab). 256 thr = 8 waves.
// LDS: s1[32][512] fp32 (64KB) + affh[32][512] f16 (32KB), dynamic.
// ---------------------------------------------------------------------------
__global__ __launch_bounds__(256) void attn_kernel(const _Float16* __restrict__ qpm, // (M, 1536)
                                                   const _Float16* __restrict__ kt,  // [n][h][l][64]
                                                   const _Float16* __restrict__ vt,  // [n][h][64][l]
                                                   const float* __restrict__ gate,   // [n][h][l]
                                                   const unsigned char* __restrict__ msk, // (N,L)
                                                   _Float16* __restrict__ mix) {     // (M, 768)
  extern __shared__ char smem[];
  float*    s1   = (float*)smem;                                  // 32*512
  _Float16* affh = (_Float16*)(smem + 32 * 512 * sizeof(float));  // 32*512
  __shared__ float redbuf[256];
  __shared__ float rowmaxS[32];
  __shared__ float rowsumS[32];

  int bx  = blockIdx.x;
  int qbi = bx & 15;            // L/32 = 16 slabs
  int h   = (bx >> 4) % NH;
  int n   = bx / (16 * NH);
  int qb0 = qbi * 32;

  int tid  = threadIdx.x;
  int wave = tid >> 5;
  int lane = tid & 31;
  int lr    = lane & 15;
  int koff  = (lane < 16) ? 0 : 8;
  int rbase = (lane < 16) ? 0 : 8;

  // ---- Pass A: S1 = q1.K^T / 8 (masked), g2 = tanh(q2.K^T/8)*gate ----
  {
    int qt  = wave & 1;   // which 16-row q tile
    int grp = wave >> 1;  // 4 groups x 8 k-tiles = 32 k-tiles (512 keys)
    const _Float16* qbase =
        qpm + (size_t)(n * LSEQ + qb0 + qt * 16 + lr) * (2 * DMODEL) + h * 128 + koff;
    v16h a1_0 = ld_frag(qbase);
    v16h a1_1 = ld_frag(qbase + 32);
    v16h a2_0 = ld_frag(qbase + 64);
    v16h a2_1 = ld_frag(qbase + 96);
    const _Float16* khead = kt + (size_t)(n * NH + h) * LSEQ * HDIM;
    for (int i = 0; i < 8; ++i) {
      int kbase = (grp * 8 + i) * 16;
      const _Float16* kp = khead + (size_t)(kbase + lr) * HDIM + koff;
      v16h b0 = ld_frag(kp);
      v16h b1 = ld_frag(kp + 32);
      v8f acc1 = {};
      v8f acc2 = {};
      acc1 = wmma16(a1_0, b0, acc1);
      acc1 = wmma16(a1_1, b1, acc1);
      acc2 = wmma16(a2_0, b0, acc2);
      acc2 = wmma16(a2_1, b1, acc2);
      int key    = kbase + lr;
      bool valid = msk[n * LSEQ + key] != 0;
      float g    = gate[(size_t)(n * NH + h) * LSEQ + key];  // 0 when masked
#pragma unroll
      for (int r8 = 0; r8 < 8; ++r8) {
        int qloc = qt * 16 + rbase + r8;
        s1[qloc * 512 + key]   = valid ? acc1[r8] * 0.125f : -__builtin_inff();
        affh[qloc * 512 + key] = (_Float16)(fast_tanh(acc2[r8] * 0.125f) * g);
      }
    }
  }
  __syncthreads();

  // ---- Pass B: softmax over keys, combine 0.5*(p + g2) -> f16 aff ----
  {
    int r  = tid & 31;  // query row within slab
    int s  = tid >> 5;  // key segment (8 x 64)
    int kb = s * 64;
    float mloc = -__builtin_inff();
    for (int j = 0; j < 64; ++j) mloc = fmaxf(mloc, s1[r * 512 + kb + j]);
    redbuf[r * 8 + s] = mloc;
    __syncthreads();
    if (s == 0) {
      float mr = redbuf[r * 8];
      for (int j = 1; j < 8; ++j) mr = fmaxf(mr, redbuf[r * 8 + j]);
      rowmaxS[r] = mr;
    }
    __syncthreads();
    float mr = rowmaxS[r];
    float sloc = 0.f;
    for (int j = 0; j < 64; ++j) {
      int idx  = r * 512 + kb + j;
      float p  = __expf(s1[idx] - mr);
      s1[idx]  = p;
      sloc    += p;
    }
    redbuf[r * 8 + s] = sloc;
    __syncthreads();
    if (s == 0) {
      float sr = 0.f;
      for (int j = 0; j < 8; ++j) sr += redbuf[r * 8 + j];
      rowsumS[r] = sr;
    }
    __syncthreads();
    float inv = fast_rcp(rowsumS[r]);
    for (int j = 0; j < 64; ++j) {
      int idx   = r * 512 + kb + j;
      float av  = 0.5f * (s1[idx] * inv + (float)affh[idx]);
      affh[idx] = (_Float16)av;
    }
  }
  __syncthreads();

  // ---- Pass C: mix(32 x 64) = aff(32 x 512) @ V(512 x 64) ----
  {
    int qt = wave & 1;
    int ct = wave >> 1;  // 4 col tiles of 16 over HD=64
    v8f acc = {};
    const _Float16* vhead =
        vt + ((size_t)(n * NH + h) * HDIM + ct * 16 + lr) * LSEQ + koff;
    const _Float16* abase = affh + (qt * 16 + lr) * 512 + koff;
    for (int kk = 0; kk < LSEQ; kk += 32) {
      v16h a = ld_frag(abase + kk);
      v16h b = ld_frag(vhead + kk);
      acc = wmma16(a, b, acc);
    }
    int col = h * HDIM + ct * 16 + lr;
#pragma unroll
    for (int r8 = 0; r8 < 8; ++r8) {
      int row = n * LSEQ + qb0 + qt * 16 + rbase + r8;
      mix[(size_t)row * DMODEL + col] = (_Float16)acc[r8];
    }
  }
}

// ---------------------------------------------------------------------------
extern "C" void kernel_launch(void* const* d_in, const int* in_sizes, int n_in,
                              void* d_out, int out_size, void* d_ws, size_t ws_size,
                              hipStream_t stream) {
  (void)in_sizes; (void)n_in; (void)out_size; (void)ws_size;
  const float* x      = (const float*)d_in[0];
  const float* kin    = (const float*)d_in[1];
  const float* vin    = (const float*)d_in[2];
  const unsigned char* m = (const unsigned char*)d_in[3];
  const float* ln1_w  = (const float*)d_in[4];
  const float* ln1_b  = (const float*)d_in[5];
  const float* in_w   = (const float*)d_in[6];
  const float* in_b   = (const float*)d_in[7];
  const float* out_w  = (const float*)d_in[8];
  const float* out_b  = (const float*)d_in[9];
  const float* ln2_w  = (const float*)d_in[10];
  const float* ln2_b  = (const float*)d_in[11];
  const float* fc_w   = (const float*)d_in[12];
  const float* fc_b   = (const float*)d_in[13];
  const float* proj_w = (const float*)d_in[14];
  const float* proj_b = (const float*)d_in[15];

  char* wsb = (char*)d_ws;
  size_t off = 0;
  auto take = [&](size_t bytes) {
    char* p = wsb + off;
    off += (bytes + 255) & ~(size_t)255;
    return (void*)p;
  };
  _Float16* qn     = (_Float16*)take((size_t)MROWS * DMODEL * 2);
  _Float16* qp     = (_Float16*)take((size_t)MROWS * 2 * DMODEL * 2);
  _Float16* inw_h  = (_Float16*)take((size_t)2 * DMODEL * DMODEL * 2);
  _Float16* outw_h = (_Float16*)take((size_t)DMODEL * DMODEL * 2);
  _Float16* fcw_h  = (_Float16*)take((size_t)4 * DMODEL * DMODEL * 2);
  _Float16* pjw_h  = (_Float16*)take((size_t)DMODEL * 4 * DMODEL * 2);
  _Float16* kt     = (_Float16*)take((size_t)NB * NH * LSEQ * HDIM * 2);
  _Float16* vt     = (_Float16*)take((size_t)NB * NH * HDIM * LSEQ * 2);
  float*    gateb  = (float*)take((size_t)NB * NH * LSEQ * 4);
  _Float16* mixh   = (_Float16*)take((size_t)MROWS * DMODEL * 2);
  float*    x1     = (float*)take((size_t)MROWS * DMODEL * 4);
  _Float16* ln2h   = (_Float16*)take((size_t)MROWS * DMODEL * 2);
  _Float16* hfc    = (_Float16*)take((size_t)MROWS * 4 * DMODEL * 2);

  // Weight / K / V conversion to f16
  cvt_f16_kernel<<<(2 * DMODEL * DMODEL + 255) / 256, 256, 0, stream>>>(in_w, inw_h, 2 * DMODEL * DMODEL);
  cvt_f16_kernel<<<(DMODEL * DMODEL + 255) / 256, 256, 0, stream>>>(out_w, outw_h, DMODEL * DMODEL);
  cvt_f16_kernel<<<(4 * DMODEL * DMODEL + 255) / 256, 256, 0, stream>>>(fc_w, fcw_h, 4 * DMODEL * DMODEL);
  cvt_f16_kernel<<<(4 * DMODEL * DMODEL + 255) / 256, 256, 0, stream>>>(proj_w, pjw_h, 4 * DMODEL * DMODEL);
  permute_k_kernel<<<(NB * LSEQ * NH * HDIM) / 256, 256, 0, stream>>>(kin, kt);
  permute_v_kernel<<<(NB * LSEQ * NH * HDIM) / 256, 256, 0, stream>>>(vin, vt);

  // LN1 -> qn (f16)
  ln_f16_kernel<<<MROWS, 256, 0, stream>>>(x, ln1_w, ln1_b, qn);

  // qp = qn @ in_w^T + in_b  (M=4096, N=1536, K=768) -> f16
  gemm_f16w<0><<<dim3((2 * DMODEL) / 256, MROWS / 64), 256, 0, stream>>>(
      qn, DMODEL, inw_h, DMODEL, in_b, nullptr, 0, qp, 2 * DMODEL, DMODEL);

  // gate precompute
  gate_kernel<<<(NB * NH * LSEQ) / 256, 256, 0, stream>>>(qp, kt, m, gateb);

  // attention -> mix (f16)
  attn_kernel<<<NB * NH * (LSEQ / 32), 256, 32 * 512 * 4 + 32 * 512 * 2, stream>>>(
      qp, kt, vt, gateb, m, mixh);

  // x1 = x + mix @ out_w^T + out_b  (N=768, K=768) -> f32
  gemm_f16w<1><<<dim3(DMODEL / 256, MROWS / 64), 256, 0, stream>>>(
      mixh, DMODEL, outw_h, DMODEL, out_b, x, DMODEL, x1, DMODEL, DMODEL);

  // LN2 -> ln2h (f16)
  ln_f16_kernel<<<MROWS, 256, 0, stream>>>(x1, ln2_w, ln2_b, ln2h);

  // hfc = silu'(ln2h @ fc_w^T + fc_b)  (N=3072, K=768) -> f16
  gemm_f16w<2><<<dim3((4 * DMODEL) / 256, MROWS / 64), 256, 0, stream>>>(
      ln2h, DMODEL, fcw_h, DMODEL, fc_b, nullptr, 0, hfc, 4 * DMODEL, DMODEL);

  // out = x1 + hfc @ proj_w^T + proj_b  (N=768, K=3072) -> f32 d_out
  gemm_f16w<1><<<dim3(DMODEL / 256, MROWS / 64), 256, 0, stream>>>(
      hfc, 4 * DMODEL, pjw_h, 4 * DMODEL, proj_b, x1, DMODEL, d_out, DMODEL, 4 * DMODEL);
}
